// DGM_28638841929775
// MI455X (gfx1250) — matrix operand order repaired and checked by reading, not compiled, not run on hardware
//
#include <hip/hip_runtime.h>
#include <hip/hip_bf16.h>
#include <math.h>

// ---------------- constants ----------------
#define CC    64
#define HH    128
#define WW    128
#define HWSZ  16384           // 128*128 = 1<<14
#define NB    2
#define NP    32768           // NB*HWSZ = 1<<15

typedef __attribute__((ext_vector_type(16))) _Float16 v16h;
typedef __attribute__((ext_vector_type(8)))  _Float16 v8h;
typedef __attribute__((ext_vector_type(8)))  float    v8f;

// ---------------- WMMA helpers ----------------
// A fragment (16x32 f16, MxK): lane l holds row m = l&15,
// element i -> K = (l>=16?16:0) + i
__device__ __forceinline__ int kOfA(int i, int l) {
  return (l & 16) + i;
}

__device__ __forceinline__ v16h ld16(const _Float16* p) {
  union { v16h v; v8h h[2]; } u;
  u.h[0] = *(const v8h*)(p);
  u.h[1] = *(const v8h*)(p + 8);
  return u.v;
}

// packed-A layout: (((mt*nkc + kc)*32 + lane)*16 + i)
__device__ __forceinline__ v16h loadA(const _Float16* __restrict__ pack,
                                      int mt, int kc, int nkc, int lane) {
  const _Float16* p = pack + ((size_t)((mt * nkc + kc) * 32 + lane) << 4);
  return ld16(p);
}

// B fragment from LDS panel laid out [n][strideF16], K-chunk starts at `base`.
// lane l -> column n = l&15, elements i -> K = (l&16?16:0)+i (16 contiguous f16)
__device__ __forceinline__ v16h loadB(const _Float16* base, int strideF16,
                                      int n0, int lane) {
  const _Float16* p = base + (size_t)(n0 + (lane & 15)) * strideF16 +
                      ((lane & 16) ? 16 : 0);
  return ld16(p);
}

__device__ __forceinline__ v8f wmma16(v16h a, v16h b, v8f c) {
  return __builtin_amdgcn_wmma_f32_16x16x32_f16(
      /*neg_a=*/false, a, /*neg_b=*/false, b,
      /*c_mod=*/(short)0, c, /*reuse_a=*/false, /*reuse_b=*/false);
}

// ---------------- prep: fp32 activations -> f16 panel [128][NP] ----------------
// rows 0..63 = input_feat, rows 64..127 = degrad_repr; column n = b*HW + hw
__global__ __launch_bounds__(256) void k0_act(const float* __restrict__ xin,
                                              const float* __restrict__ xdg,
                                              _Float16* __restrict__ fea16) {
  for (int e = 0; e < 4; ++e) {
    size_t i = (size_t)blockIdx.x * 1024 + threadIdx.x + 256 * e;
    int r  = (int)(i >> 15);
    int n  = (int)(i & (NP - 1));
    int b  = n >> 14;
    int hw = n & (HWSZ - 1);
    int c  = r & 63;
    const float* src = (r < 64) ? xin : xdg;
    fea16[i] = (_Float16)src[((size_t)(b * CC + c) << 14) + hw];
  }
}

// ---------------- prep: pack weights into A-fragment layout ----------------
// mode 0: W1cat  [128x64]   rows<64: g_w1, else b_w1
// mode 1: W2bd   [128x128]  block-diag g_w2 / b_w2
// mode 2: OffW   [32x1152]  rows<27: off_w[m][c][t], k=c*9+t
// mode 3: DcnW   [64x576]   dcn_w[m][c][t], k=c*9+t
__global__ __launch_bounds__(256) void k_pack(const float* __restrict__ p0,
                                              const float* __restrict__ p1,
                                              _Float16* __restrict__ dst,
                                              int mode, int nkc) {
  int idx = blockIdx.x * 256 + threadIdx.x;
  int i  = idx & 15;
  int l  = (idx >> 4) & 31;
  int mk = idx >> 9;
  int kc = mk % nkc;
  int mt = mk / nkc;
  int m  = mt * 16 + (l & 15);
  int k  = kc * 32 + kOfA(i, l);
  float v = 0.f;
  if (mode == 0) {
    v = (m < 64) ? p0[m * 64 + k] : p1[(m - 64) * 64 + k];
  } else if (mode == 1) {
    if (m < 64) { if (k < 64)  v = p0[m * 64 + k]; }
    else        { if (k >= 64) v = p1[(m - 64) * 64 + (k - 64)]; }
  } else if (mode == 2) {
    if (m < 27) { int c = k / 9, t = k - c * 9; v = p0[(m * 128 + c) * 9 + t]; }
  } else {
    int c = k / 9, t = k - c * 9;
    v = p0[(m * 64 + c) * 9 + t];
  }
  dst[idx] = (_Float16)v;
}

// ---------------- K1: T12 = lrelu(W1cat @ Xdg + b) ; M=128,K=64,Ntile=64 ----------------
__global__ __launch_bounds__(256) void k1_sft_hidden(
    const _Float16* __restrict__ fea16, const float* __restrict__ gb1,
    const float* __restrict__ bb1, _Float16* __restrict__ t12,
    const _Float16* __restrict__ packW1) {
  __shared__ _Float16 sB[64 * 40];
  const int tid = threadIdx.x, lane = tid & 31, wave = tid >> 5;
  const int n0 = blockIdx.x * 64;
  const _Float16* xdg = fea16 + (size_t)64 * NP;   // degrad rows

  v8f acc0 = {}, acc1 = {}, acc2 = {}, acc3 = {};
  for (int kc = 0; kc < 2; ++kc) {
    for (int e = 0; e < 8; ++e) {
      int idx = tid + 256 * e;                 // 2048 = 32k x 64n
      int nn = idx & 63, kk = idx >> 6;
      sB[nn * 40 + kk] = xdg[(size_t)(kc * 32 + kk) * NP + (n0 + nn)];
    }
    if (kc + 1 < 2 && tid < 32)                // prefetch next K-chunk rows
      __builtin_prefetch(&xdg[(size_t)((kc + 1) * 32 + tid) * NP + n0], 0, 3);
    __syncthreads();
    v16h a = loadA(packW1, wave, kc, 2, lane);
    acc0 = wmma16(a, loadB(sB, 40,  0, lane), acc0);
    acc1 = wmma16(a, loadB(sB, 40, 16, lane), acc1);
    acc2 = wmma16(a, loadB(sB, 40, 32, lane), acc2);
    acc3 = wmma16(a, loadB(sB, 40, 48, lane), acc3);
    __syncthreads();
  }
  int mbase = wave * 16 + ((lane & 16) ? 8 : 0);
  v8f accs[4] = {acc0, acc1, acc2, acc3};
  for (int s = 0; s < 4; ++s) {
    int n = n0 + s * 16 + (lane & 15);
    for (int j = 0; j < 8; ++j) {
      int m = mbase + j;
      float bias = (m < 64) ? gb1[m] : bb1[m - 64];
      float v = accs[s][j] + bias;
      v = (v > 0.f) ? v : 0.1f * v;
      t12[(size_t)m * NP + n] = (_Float16)v;
    }
  }
}

// ---------------- K2: gamma/beta = W2bd @ T12 ; out = x*g + b + x ; Ntile=64 ----------------
__global__ __launch_bounds__(256) void k2_sft_out(
    const _Float16* __restrict__ t12, const float* __restrict__ xin,
    const float* __restrict__ gb2, const float* __restrict__ bb2,
    float* __restrict__ out, const _Float16* __restrict__ packW2) {
  __shared__ _Float16 sB[64 * 40];
  __shared__ float sGB[128 * 64];
  const int tid = threadIdx.x, lane = tid & 31, wave = tid >> 5;
  const int n0 = blockIdx.x * 64;

  v8f acc0 = {}, acc1 = {}, acc2 = {}, acc3 = {};
  for (int kc = 0; kc < 4; ++kc) {
    for (int e = 0; e < 8; ++e) {
      int idx = tid + 256 * e;                 // 2048 = 32k x 64n
      int nn = idx & 63, kk = idx >> 6;
      sB[nn * 40 + kk] = t12[(size_t)(kc * 32 + kk) * NP + (n0 + nn)];
    }
    if (kc + 1 < 4 && tid < 32)                // prefetch next K-chunk rows
      __builtin_prefetch(&t12[(size_t)((kc + 1) * 32 + tid) * NP + n0], 0, 3);
    __syncthreads();
    v16h a = loadA(packW2, wave, kc, 4, lane);
    acc0 = wmma16(a, loadB(sB, 40,  0, lane), acc0);
    acc1 = wmma16(a, loadB(sB, 40, 16, lane), acc1);
    acc2 = wmma16(a, loadB(sB, 40, 32, lane), acc2);
    acc3 = wmma16(a, loadB(sB, 40, 48, lane), acc3);
    __syncthreads();
  }
  int mbase = wave * 16 + ((lane & 16) ? 8 : 0);
  int nn = lane & 15;
  v8f accs[4] = {acc0, acc1, acc2, acc3};
  for (int s = 0; s < 4; ++s)
    for (int j = 0; j < 8; ++j)
      sGB[(mbase + j) * 64 + s * 16 + nn] = accs[s][j];
  __syncthreads();
  for (int e = 0; e < 16; ++e) {
    int idx = tid + 256 * e;                   // 4096 = 64c x 64n
    int n_ = idx & 63, c = idx >> 6;
    float gamma = sGB[c * 64 + n_] + gb2[c];
    float beta  = sGB[(64 + c) * 64 + n_] + bb2[c];
    int ng = n0 + n_;
    int b  = ng >> 14;
    int hw = ng & (HWSZ - 1);
    size_t xi = ((size_t)(b * CC + c) << 14) + hw;
    float x = xin[xi];
    out[xi] = x * gamma + beta + x;
  }
}

// ---------------- K3: offset conv (implicit GEMM) M=32,K=1152,Ntile=64 ----------------
__global__ __launch_bounds__(256) void k3_offconv(
    const _Float16* __restrict__ fea16, const float* __restrict__ offb,
    float* __restrict__ om, const _Float16* __restrict__ packWo) {
  __shared__ _Float16 sB[64 * 40];
  const int tid = threadIdx.x, lane = tid & 31, wave = tid >> 5;
  const int mt = wave & 1, nt = wave >> 1;
  const int n0 = blockIdx.x * 64;
  const int b = n0 >> 14, hwb = n0 & (HWSZ - 1);

  v8f acc = {};
  for (int kc = 0; kc < 36; ++kc) {
    for (int e = 0; e < 8; ++e) {
      int idx = tid + 256 * e;                 // 2048 = 32k x 64n
      int nn = idx & 63, kk = idx >> 6;
      int k = kc * 32 + kk;
      int c = k / 9, t = k - c * 9;
      int hw = hwb + nn;
      int h = hw >> 7, w = hw & 127;
      int sy = h + t / 3 - 1;
      int sx = w + (t % 3) - 1;
      _Float16 v = (_Float16)0.f;
      if (sy >= 0 && sy < HH && sx >= 0 && sx < WW)
        v = fea16[(size_t)c * NP + (b << 14) + (sy << 7) + sx];
      sB[nn * 40 + kk] = v;
    }
    if (kc + 1 < 36 && tid < 32) {             // prefetch next chunk's channel rows
      int k = (kc + 1) * 32 + tid;
      int c = k / 9;
      __builtin_prefetch(&fea16[(size_t)c * NP + (b << 14) + hwb], 0, 3);
    }
    __syncthreads();
    v16h a = loadA(packWo, mt, kc, 36, lane);
    v16h bf = loadB(sB, 40, nt * 16, lane);
    acc = wmma16(a, bf, acc);
    __syncthreads();
  }
  int mbase = mt * 16 + ((lane & 16) ? 8 : 0);
  int n = n0 + nt * 16 + (lane & 15);
  int hw = n & (HWSZ - 1);
  for (int j = 0; j < 8; ++j) {
    int m = mbase + j;
    if (m < 27)
      om[((size_t)(b * 32 + m) << 14) + hw] = acc[j] + offb[m];
  }
}

// ---------------- K4: modulated deform conv; gather -> LDS -> GEMM ----------------
__global__ __launch_bounds__(256) void k4_dcn(
    const float* __restrict__ xin, const float* __restrict__ om,
    const float* __restrict__ dcnb, float* __restrict__ out,
    const _Float16* __restrict__ packWd) {
  __shared__ int   sY0[288], sX0[288];
  __shared__ float sWy[288], sWx[288], sMsk[288];
  __shared__ _Float16 sVal[32 * 584];          // [n][K=576 (+pad)]
  const int tid = threadIdx.x, lane = tid & 31, wave = tid >> 5;
  const int n0 = blockIdx.x * 32;
  const int b = n0 >> 14, hw0 = n0 & (HWSZ - 1);

  // per-(pixel,tap) bilinear params
  for (int i = tid; i < 288; i += 256) {
    int nn = i / 9, t = i - nn * 9;
    int hw = hw0 + nn;
    int h = hw >> 7, w = hw & 127;
    size_t ob = ((size_t)(b * 32) << 14) + hw;
    float dy = om[ob + ((size_t)t << 14)];
    float dx = om[ob + ((size_t)(9 + t) << 14)];
    float mr = om[ob + ((size_t)(18 + t) << 14)];
    float py = (float)h + (float)(t / 3 - 1) + dy;
    float px = (float)w + (float)(t % 3 - 1) + dx;
    float fy = floorf(py), fx = floorf(px);
    sY0[i] = (int)fy; sX0[i] = (int)fx;
    sWy[i] = py - fy; sWx[i] = px - fx;
    sMsk[i] = 1.f / (1.f + expf(-mr));
  }
  __syncthreads();

  // gather 64ch x 9taps x 32px, modulated, into f16 B panel
  for (int t = 0; t < 9; ++t) {
    for (int e = 0; e < 8; ++e) {
      int idx = tid + 256 * e;                 // 2048 = 64c x 32n
      int nn = idx & 31, c = idx >> 5;
      int pi = nn * 9 + t;
      int y0 = sY0[pi], x0 = sX0[pi];
      float wy1 = sWy[pi], wx1 = sWx[pi];
      float wy0 = 1.f - wy1, wx0 = 1.f - wx1;
      const float* xp = xin + (((size_t)(b * CC + c)) << 14);
      float v00 = 0.f, v01 = 0.f, v10 = 0.f, v11 = 0.f;
      bool y0v = (y0 >= 0) && (y0 < HH);
      bool y1v = (y0 + 1 >= 0) && (y0 + 1 < HH);
      bool x0v = (x0 >= 0) && (x0 < WW);
      bool x1v = (x0 + 1 >= 0) && (x0 + 1 < WW);
      if (y0v && x0v) v00 = xp[(y0 << 7) + x0];
      if (y0v && x1v) v01 = xp[(y0 << 7) + x0 + 1];
      if (y1v && x0v) v10 = xp[((y0 + 1) << 7) + x0];
      if (y1v && x1v) v11 = xp[((y0 + 1) << 7) + x0 + 1];
      float v = (v00 * wx0 + v01 * wx1) * wy0 + (v10 * wx0 + v11 * wx1) * wy1;
      v *= sMsk[pi];
      sVal[nn * 584 + c * 9 + t] = (_Float16)v;
    }
  }
  __syncthreads();

  // GEMM: M=64 (4 mtiles) x N=32 (2 ntiles), K=576 (18 chunks of 32)
  const int mt = wave & 3, nt = wave >> 2;
  v8f acc = {};
  for (int kc = 0; kc < 18; ++kc) {
    v16h a = loadA(packWd, mt, kc, 18, lane);
    v16h bf = loadB(sVal + kc * 32, 584, nt * 16, lane);
    acc = wmma16(a, bf, acc);
  }
  int mbase = mt * 16 + ((lane & 16) ? 8 : 0);
  int n = n0 + nt * 16 + (lane & 15);
  int hw = n & (HWSZ - 1);
  for (int j = 0; j < 8; ++j) {
    int m = mbase + j;
    size_t oi = (((size_t)(b * CC + m)) << 14) + hw;
    out[oi] += acc[j] + dcnb[m];
  }
}

// ---------------- host launcher ----------------
extern "C" void kernel_launch(void* const* d_in, const int* in_sizes, int n_in,
                              void* d_out, int out_size, void* d_ws, size_t ws_size,
                              hipStream_t stream) {
  (void)in_sizes; (void)n_in; (void)out_size; (void)ws_size;
  const float* in_feat = (const float*)d_in[0];
  const float* degrad  = (const float*)d_in[1];
  const float* g_w1 = (const float*)d_in[2];
  const float* g_b1 = (const float*)d_in[3];
  const float* g_w2 = (const float*)d_in[4];
  const float* g_b2 = (const float*)d_in[5];
  const float* b_w1 = (const float*)d_in[6];
  const float* b_b1 = (const float*)d_in[7];
  const float* b_w2 = (const float*)d_in[8];
  const float* b_b2 = (const float*)d_in[9];
  const float* off_w = (const float*)d_in[10];
  const float* off_b = (const float*)d_in[11];
  const float* dcn_w = (const float*)d_in[12];
  const float* dcn_b = (const float*)d_in[13];
  float* out = (float*)d_out;

  char* ws = (char*)d_ws;
  _Float16* fea16 = (_Float16*)(ws + 0);                 // 128*NP f16 = 8 MB
  _Float16* t12   = (_Float16*)(ws + (size_t)8388608);   // 128*NP f16 = 8 MB
  float*    om    = (float*)   (ws + (size_t)16777216);  // 2*32*HW f32 = 4 MB
  _Float16* pw1   = (_Float16*)(ws + (size_t)20971520);  // 8192 f16
  _Float16* pw2   = (_Float16*)(ws + (size_t)20987904);  // 16384 f16
  _Float16* pwo   = (_Float16*)(ws + (size_t)21020672);  // 36864 f16
  _Float16* pwd   = (_Float16*)(ws + (size_t)21094400);  // 36864 f16

  // prep
  k0_act<<<4096, 256, 0, stream>>>(in_feat, degrad, fea16);
  k_pack<<<32,  256, 0, stream>>>(g_w1, b_w1, pw1, 0, 2);      // 8mt*2kc
  k_pack<<<64,  256, 0, stream>>>(g_w2, b_w2, pw2, 1, 4);      // 8mt*4kc
  k_pack<<<144, 256, 0, stream>>>(off_w, nullptr, pwo, 2, 36); // 2mt*36kc
  k_pack<<<144, 256, 0, stream>>>(dcn_w, nullptr, pwd, 3, 18); // 4mt*18kc

  // SFT chain
  k1_sft_hidden<<<NP / 64, 256, 0, stream>>>(fea16, g_b1, b_b1, t12, pw1);
  k2_sft_out  <<<NP / 64, 256, 0, stream>>>(t12, in_feat, g_b2, b_b2, out, pw2);

  // offset conv + deformable conv
  k3_offconv<<<NP / 64, 256, 0, stream>>>(fea16, off_b, om, pwo);
  k4_dcn    <<<NP / 32, 256, 0, stream>>>(in_feat, om, dcn_b, out, pwd);
}
